// VariationalGATEncoder_74028056314068
// MI455X (gfx1250) — matrix operand order repaired and checked by reading, not compiled, not run on hardware
//
#include <hip/hip_runtime.h>
#include <hip/hip_bf16.h>

// ---------------- problem constants ----------------
#define NN   50000      // nodes
#define EE   800000     // edges
#define FIN  128        // input width to every GEMM in this model
#define MT   (NN / 16)  // 3125 M-tiles (exact)
#define KT   (FIN / 32) // 4 K-tiles (exact)
#define NEG_SLOPE 0.2f
#define EPS_F 1e-16f

typedef __attribute__((ext_vector_type(16))) _Float16 v16h;
typedef __attribute__((ext_vector_type(8)))  float    v8f;

// ---------------- tiny utility kernels ----------------
__global__ void fill_kernel(float* __restrict__ p, unsigned long long n, float v) {
    unsigned long long i = (unsigned long long)blockIdx.x * blockDim.x + threadIdx.x;
    if (i < n) p[i] = v;
}

// d_out init: mu half = N*bmu, logvar half = N*blv (bias folded into column sum)
__global__ void out_init_kernel(float* __restrict__ out, const float* __restrict__ bmu,
                                const float* __restrict__ blv) {
    int c = threadIdx.x;
    if (c < 64)       out[c] = (float)NN * bmu[c];
    else if (c < 128) out[c] = (float)NN * blv[c - 64];
}

__global__ void bias_relu_kernel(float* __restrict__ h, const float* __restrict__ b, int F) {
    unsigned long long i = (unsigned long long)blockIdx.x * blockDim.x + threadIdx.x;
    if (i >= (unsigned long long)NN * F) return;
    float v = h[i] + b[i % F];
    h[i] = v > 0.f ? v : 0.f;
}

// ---------------- WMMA tile packing (ISA 7.12.2 layouts) ----------------
// A tile 16x32 f16: lanes 0-15 -> M=lane, K: vgpr0..3 -> K=2d+half, vgpr4..7 -> K=16+2(d-4)+half
//                   lanes 16-31 -> M=lane-16, K group offset +8
__global__ void pack_a_kernel(const float* __restrict__ X, _Float16* __restrict__ pA) {
    unsigned long long idx = (unsigned long long)blockIdx.x * blockDim.x + threadIdx.x;
    if (idx >= (unsigned long long)MT * KT * 256ull) return;
    int d    = (int)(idx & 7);          // dword 0..7 within lane
    int lane = (int)((idx >> 3) & 31);
    unsigned long long tile = idx >> 8; // tm*KT + tk
    int tk = (int)(tile % KT);
    int tm = (int)(tile / KT);
    int row = tm * 16 + (lane & 15);
    int k   = tk * 32 + ((d & 4) ? 16 : 0) + ((lane & 16) ? 8 : 0) + (d & 3) * 2;
    const float* src = X + (unsigned long long)row * FIN + k;
    _Float16* dst = pA + tile * 512ull + (unsigned long long)lane * 16 + d * 2;
    dst[0] = (_Float16)src[0];
    dst[1] = (_Float16)src[1];
}

// B tile 32x16 f16: lane -> col N=(lane&15); vgpr0..3 -> K=2d+half, vgpr4..7 -> K=16+2(d-4)+half;
// lanes 16-31: K group offset +8
__global__ void pack_b_kernel(const float* __restrict__ W, _Float16* __restrict__ pB,
                              int Ntiles, int Fout) {
    int idx = blockIdx.x * blockDim.x + threadIdx.x;
    if (idx >= KT * Ntiles * 256) return;
    int d    = idx & 7;
    int lane = (idx >> 3) & 31;
    int tile = idx >> 8;                // tk*Ntiles + tn
    int tn = tile % Ntiles;
    int tk = tile / Ntiles;
    int col = tn * 16 + (lane & 15);
    int k   = tk * 32 + ((d & 4) ? 16 : 0) + ((lane & 16) ? 8 : 0) + (d & 3) * 2;
    _Float16* dst = pB + (unsigned long long)tile * 512ull + (unsigned long long)lane * 16 + d * 2;
    dst[0] = (_Float16)W[(unsigned long long)k       * Fout + col];
    dst[1] = (_Float16)W[(unsigned long long)(k + 1) * Fout + col];
}

// ---------------- WMMA GEMM: H[N x Fout] = X @ W (f16 in, f32 acc, f16 out) ----------------
__global__ void gemm_wmma_kernel(const _Float16* __restrict__ pA, const _Float16* __restrict__ pB,
                                 _Float16* __restrict__ H, int Ntiles, int Fout) {
    int wave = (int)(((unsigned long long)blockIdx.x * blockDim.x + threadIdx.x) >> 5);
    int lane = threadIdx.x & 31;
    if (wave >= MT * Ntiles) return;      // wave-uniform exit, EXEC all-1 inside WMMA
    int tm = wave / Ntiles;
    int tn = wave % Ntiles;
    v8f acc = {};
#pragma unroll
    for (int tk = 0; tk < KT; ++tk) {
        v16h a = *(const v16h*)(pA + ((unsigned long long)(tm * KT + tk) * 512ull) + lane * 16);
        v16h b = *(const v16h*)(pB + ((unsigned long long)(tk * Ntiles + tn) * 512ull) + lane * 16);
        acc = __builtin_amdgcn_wmma_f32_16x16x32_f16(false, a, false, b, (short)0, acc, false, false);
    }
    // C layout: lanes 0-15: M = r, N = lane; lanes 16-31: M = 8+r, N = lane-16
    int rowBase = tm * 16 + ((lane & 16) ? 8 : 0);
    int col     = tn * 16 + (lane & 15);
#pragma unroll
    for (int r = 0; r < 8; ++r)
        H[(unsigned long long)(rowBase + r) * Fout + col] = (_Float16)acc[r];
}

// ---------------- attention logits per node: al_s = h.a_src, al_d = h.a_dst ----------------
__global__ void att_dots_kernel(const _Float16* __restrict__ H, const float* __restrict__ a_s,
                                const float* __restrict__ a_d, float* __restrict__ als,
                                float* __restrict__ ald, int Fout) {
    int wave = (int)(((unsigned long long)blockIdx.x * blockDim.x + threadIdx.x) >> 5);
    int lane = threadIdx.x & 31;
    if (wave >= NN) return;
    float s = 0.f, d = 0.f;
    for (int c = lane; c < Fout; c += 32) {
        float hv = (float)H[(unsigned long long)wave * Fout + c];
        s += hv * a_s[c];
        d += hv * a_d[c];
    }
#pragma unroll
    for (int off = 16; off > 0; off >>= 1) {
        s += __shfl_down(s, off, 32);
        d += __shfl_down(d, off, 32);
    }
    if (lane == 0) { als[wave] = s; ald[wave] = d; }
}

// ---------------- edge-level segment softmax pieces ----------------
__device__ __forceinline__ void atomicMaxFloat(float* addr, float val) {
    if (val >= 0.f) atomicMax((int*)addr, __float_as_int(val));
    else            atomicMin((unsigned int*)addr, __float_as_uint(val));
}

__global__ void edge_logits_kernel(const int* __restrict__ src, const int* __restrict__ dst,
                                   const float* __restrict__ als, const float* __restrict__ ald,
                                   float* __restrict__ ex, float* __restrict__ m) {
    int e = blockIdx.x * blockDim.x + threadIdx.x;
    if (e >= EE) return;
    int d = dst[e];
    float v = als[src[e]] + ald[d];
    v = v >= 0.f ? v : NEG_SLOPE * v;
    ex[e] = v;
    atomicMaxFloat(&m[d], v);
}

__global__ void edge_exp_kernel(const int* __restrict__ dst, float* __restrict__ ex,
                                const float* __restrict__ m, float* __restrict__ denom) {
    int e = blockIdx.x * blockDim.x + threadIdx.x;
    if (e >= EE) return;
    int d = dst[e];
    float v = __expf(ex[e] - m[d]);
    ex[e] = v;
    atomicAdd(&denom[d], v);
}

// ---------------- weighted message scatter: out[dst] += alpha * h[src] ----------------
__global__ void scatter_kernel(const int* __restrict__ src, const int* __restrict__ dst,
                               const float* __restrict__ ex, const float* __restrict__ denom,
                               const _Float16* __restrict__ H, float* __restrict__ out, int Fout) {
    int wave = (int)(((unsigned long long)blockIdx.x * blockDim.x + threadIdx.x) >> 5);
    int lane = threadIdx.x & 31;
    if (wave >= EE) return;
    int s = src[wave];
    int d = dst[wave];
    float alpha = ex[wave] / (denom[d] + EPS_F);
    const _Float16* hrow = H + (unsigned long long)s * Fout;
    float* orow = out + (unsigned long long)d * Fout;
    for (int c = lane; c < Fout; c += 32)
        atomicAdd(&orow[c], alpha * (float)hrow[c]);
}

// ---------------- column sum (for mu / log_var), bias already in res ----------------
__global__ void colsum_kernel(const float* __restrict__ A, float* __restrict__ res, int F) {
    int col  = threadIdx.x % F;
    int rpb  = blockDim.x / F;                       // rows per block per sweep
    int r0   = blockIdx.x * rpb + threadIdx.x / F;
    int step = gridDim.x * rpb;
    float acc = 0.f;
    for (int r = r0; r < NN; r += step)
        acc += A[(unsigned long long)r * F + col];
    atomicAdd(&res[col], acc);
}

// ---------------- host side ----------------
static inline unsigned long long alignUp(unsigned long long x) { return (x + 255ull) & ~255ull; }

extern "C" void kernel_launch(void* const* d_in, const int* in_sizes, int n_in,
                              void* d_out, int out_size, void* d_ws, size_t ws_size,
                              hipStream_t stream) {
    (void)in_sizes; (void)n_in; (void)out_size; (void)ws_size;

    const float* x    = (const float*)d_in[0];
    const int*   ei   = (const int*)d_in[1];
    const int*   srcI = ei;            // edge_index[0]
    const int*   dstI = ei + EE;       // edge_index[1]
    const float* W1  = (const float*)d_in[2];
    const float* as1 = (const float*)d_in[3];
    const float* ad1 = (const float*)d_in[4];
    const float* b1  = (const float*)d_in[5];
    const float* W2  = (const float*)d_in[6];
    const float* as2 = (const float*)d_in[7];
    const float* ad2 = (const float*)d_in[8];
    const float* b2  = (const float*)d_in[9];
    const float* Wmu  = (const float*)d_in[10];
    const float* asmu = (const float*)d_in[11];
    const float* admu = (const float*)d_in[12];
    const float* bmu  = (const float*)d_in[13];
    const float* Wlv  = (const float*)d_in[14];
    const float* aslv = (const float*)d_in[15];
    const float* adlv = (const float*)d_in[16];
    const float* blv  = (const float*)d_in[17];
    float* outv = (float*)d_out;

    // ---- workspace carving ----
    char* ws = (char*)d_ws;
    unsigned long long off = 0;
    auto carve = [&](unsigned long long bytes) -> void* {
        void* p = ws + off;
        off = alignUp(off + bytes);
        return p;
    };
    _Float16* hbuf  = (_Float16*)carve((unsigned long long)NN * 128 * 2);  // h (f16 messages)
    float*    buf1  = (float*)carve((unsigned long long)NN * 128 * 4);     // ping (f32 activations)
    float*    buf2  = (float*)carve((unsigned long long)NN * 128 * 4);     // pong (f32 activations)
    _Float16* pA    = (_Float16*)carve((unsigned long long)MT * KT * 512 * 2);
    _Float16* pB    = (_Float16*)carve((unsigned long long)KT * 8 * 512 * 2);
    float*    als   = (float*)carve((unsigned long long)NN * 4);
    float*    ald   = (float*)carve((unsigned long long)NN * 4);
    float*    mbuf  = (float*)carve((unsigned long long)NN * 4);
    float*    denom = (float*)carve((unsigned long long)NN * 4);
    float*    exbuf = (float*)carve((unsigned long long)EE * 4);

    const int TPB = 256;
    auto blocksFor = [](unsigned long long n, int tpb) -> unsigned int {
        return (unsigned int)((n + tpb - 1) / tpb);
    };

    // one full GAT conv: in (N x 128, f32) --> outbuf (N x Fout, f32), NO bias/relu applied
    auto run_gat = [&](const float* in, const float* W, const float* a_s, const float* a_d,
                       float* outbuf, int Fout) {
        int Ntiles = Fout / 16;
        // init per-layer accumulators
        fill_kernel<<<blocksFor(NN, TPB), TPB, 0, stream>>>(mbuf, NN, -3.402823466e38f);
        fill_kernel<<<blocksFor(NN, TPB), TPB, 0, stream>>>(denom, NN, 0.f);
        fill_kernel<<<blocksFor((unsigned long long)NN * Fout, TPB), TPB, 0, stream>>>(
            outbuf, (unsigned long long)NN * Fout, 0.f);
        // pack + WMMA GEMM: h = in @ W  (h stored f16)
        pack_a_kernel<<<blocksFor((unsigned long long)MT * KT * 256, TPB), TPB, 0, stream>>>(in, pA);
        pack_b_kernel<<<blocksFor((unsigned long long)KT * Ntiles * 256, TPB), TPB, 0, stream>>>(
            W, pB, Ntiles, Fout);
        gemm_wmma_kernel<<<blocksFor((unsigned long long)MT * Ntiles * 32, TPB), TPB, 0, stream>>>(
            pA, pB, hbuf, Ntiles, Fout);
        // attention logits per node
        att_dots_kernel<<<blocksFor((unsigned long long)NN * 32, TPB), TPB, 0, stream>>>(
            hbuf, a_s, a_d, als, ald, Fout);
        // segment softmax over destination neighborhoods
        edge_logits_kernel<<<blocksFor(EE, TPB), TPB, 0, stream>>>(srcI, dstI, als, ald, exbuf, mbuf);
        edge_exp_kernel<<<blocksFor(EE, TPB), TPB, 0, stream>>>(dstI, exbuf, mbuf, denom);
        // weighted message scatter-add
        scatter_kernel<<<blocksFor((unsigned long long)EE * 32, TPB), TPB, 0, stream>>>(
            srcI, dstI, exbuf, denom, hbuf, outbuf, Fout);
    };

    // layer 1: x -> buf1, relu(+b1)
    run_gat(x, W1, as1, ad1, buf1, 128);
    bias_relu_kernel<<<blocksFor((unsigned long long)NN * 128, TPB), TPB, 0, stream>>>(buf1, b1, 128);
    // layer 2 (shared conv2), repeat 1: buf1 -> buf2
    run_gat(buf1, W2, as2, ad2, buf2, 128);
    bias_relu_kernel<<<blocksFor((unsigned long long)NN * 128, TPB), TPB, 0, stream>>>(buf2, b2, 128);
    // layer 2 repeat 2: buf2 -> buf1
    run_gat(buf2, W2, as2, ad2, buf1, 128);
    bias_relu_kernel<<<blocksFor((unsigned long long)NN * 128, TPB), TPB, 0, stream>>>(buf1, b2, 128);

    // init output (bias folded: sum_n (row + b) = colsum + N*b)
    out_init_kernel<<<1, 128, 0, stream>>>(outv, bmu, blv);

    // mu head: buf1 -> buf2 (N x 64), column-sum into outv[0:64]
    run_gat(buf1, Wmu, asmu, admu, buf2, 64);
    colsum_kernel<<<256, 256, 0, stream>>>(buf2, outv, 64);

    // log_var head: buf1 -> buf2 (N x 64), column-sum into outv[64:128]
    run_gat(buf1, Wlv, aslv, adlv, buf2, 64);
    colsum_kernel<<<256, 256, 0, stream>>>(buf2, outv + 64, 64);
}
